// CrossNetwork_91242285237053
// MI455X (gfx1250) — matrix operand order repaired and checked by reading, not compiled, not run on hardware
//
#include <hip/hip_runtime.h>

#define B_N 4096
#define F_N 64
#define E_N 128
#define T_N 2

typedef __attribute__((ext_vector_type(8)))  float  v8f;
typedef __attribute__((ext_vector_type(8)))  __bf16 v8bf;
typedef __attribute__((ext_vector_type(16))) __bf16 v16bf;

// ---- WMMA helpers -----------------------------------------------------------

__device__ __forceinline__ v8f wmma_bf16(v16bf a, v16bf b, v8f c) {
  // D = A(16x32 bf16) x B(32x16 bf16) + C(f32)
  return __builtin_amdgcn_wmma_f32_16x16x32_bf16(
      /*neg_a=*/false, a, /*neg_b=*/false, b,
      /*c_mod=*/(short)0, c, /*reuse_a=*/false, /*reuse_b=*/false);
}

// A-matrix 16x32 bf16 fragment: lane half hf holds 16 consecutive K
// {k0+hf*16..k0+hf*16+15}, matching ldB. Two 16B chunks.
__device__ __forceinline__ v16bf ldA(const __bf16* rowp, int k0, int hf) {
  v8bf a0 = *(const v8bf*)(rowp + k0 + hf * 16);
  v8bf a1 = *(const v8bf*)(rowp + k0 + hf * 16 + 8);
  return __builtin_shufflevector(a0, a1, 0, 1, 2, 3, 4, 5, 6, 7,
                                 8, 9, 10, 11, 12, 13, 14, 15);
}

// B-matrix 32x16 bf16 fragment: per lane 16 consecutive K starting at k0+hf*16
// for column n = lane&15.  p must point at that run (16B aligned).
__device__ __forceinline__ v16bf ldB(const __bf16* p) {
  v8bf b0 = *(const v8bf*)(p);
  v8bf b1 = *(const v8bf*)(p + 8);
  return __builtin_shufflevector(b0, b1, 0, 1, 2, 3, 4, 5, 6, 7,
                                 8, 9, 10, 11, 12, 13, 14, 15);
}

__device__ __forceinline__ void split_bf16(float v, __bf16* hi, __bf16* lo) {
  __bf16 h = (__bf16)v;
  *hi = h;
  *lo = (__bf16)(v - (float)h);
}

// ---- K0: adjacency ----------------------------------------------------------
// adj[t,f,g] = relu(m0*m1); row-normalize by (sum+1e-6). grid=128 blocks of 64.
__global__ __launch_bounds__(64) void k_adj(const float* __restrict__ masker,
                                            float* __restrict__ adjg) {
  __shared__ float red[64];
  const int t = blockIdx.x >> 6;
  const int f = blockIdx.x & 63;
  const int g = threadIdx.x;
  float m0 = masker[t * 8192 + f * 64 + g];
  float m1 = masker[t * 8192 + 4096 + f * 64 + g];
  float v = m0 * m1;
  v = v > 0.f ? v : 0.f;
  red[g] = v;
  __syncthreads();
  for (int st = 32; st > 0; st >>= 1) {
    if (g < st) red[g] += red[g + st];
    __syncthreads();
  }
  adjg[t * 4096 + f * 64 + g] = v / (red[0] + 1e-6f);
}

// ---- K1: fused split-bf16 WMMA GEMMs ---------------------------------------
// One block (8 waves) per sample b. 224KB dynamic LDS.
__global__ __launch_bounds__(256) void k_gemm(
    const float* __restrict__ x, const float* __restrict__ w_lin,
    const float* __restrict__ b_lin, const float* __restrict__ w_agg,
    const float* __restrict__ b_agg, const float* __restrict__ adjg,
    float* __restrict__ outf) {
  extern __shared__ char smemc[];
  __bf16* xh   = (__bf16*)(smemc);            // [64][128] 16KB
  __bf16* xl   = (__bf16*)(smemc + 16384);
  __bf16* wlh  = (__bf16*)(smemc + 32768);    // [o=128][e=128] 32KB
  __bf16* wll  = (__bf16*)(smemc + 65536);
  __bf16* wah  = (__bf16*)(smemc + 98304);
  __bf16* wal  = (__bf16*)(smemc + 131072);
  __bf16* adjh = (__bf16*)(smemc + 163840);   // [2][64][64] 16KB
  __bf16* adjl = (__bf16*)(smemc + 180224);
  __bf16* y2h  = (__bf16*)(smemc + 196608);   // transposed [e=128][g=64] 16KB
  __bf16* y2l  = (__bf16*)(smemc + 212992);

  const int b = blockIdx.x;
  const int tid = threadIdx.x;

  const float* xb = x + (size_t)b * (F_N * E_N);
  for (int i = tid; i < F_N * E_N; i += 256) split_bf16(xb[i], &xh[i], &xl[i]);
  for (int i = tid; i < E_N * E_N; i += 256) {
    split_bf16(w_lin[i], &wlh[i], &wll[i]);
    split_bf16(w_agg[i], &wah[i], &wal[i]);
  }
  for (int i = tid; i < T_N * F_N * F_N; i += 256)
    split_bf16(adjg[i], &adjh[i], &adjl[i]);
  __syncthreads();

  const int wave = tid >> 5;
  const int lane = tid & 31;
  const int hf = lane >> 4;   // half-wave
  const int lr = lane & 15;
  const int e0 = wave * 16;   // this wave owns output columns e0..e0+15

  v8f y1acc[4], y2acc[4];
#pragma unroll
  for (int ft = 0; ft < 4; ++ft) {
    y1acc[ft] = (v8f){};
    y2acc[ft] = (v8f){};
  }

  // Stage 1: y1 = x@Wl^T, y2 = x@Wa^T   (K = 128, 4 steps of 32)
  for (int k0 = 0; k0 < E_N; k0 += 32) {
    const int wb = (e0 + lr) * E_N + k0 + hf * 16;  // B-frag: row o, 16 consec e
    v16bf Blh = ldB(wlh + wb);
    v16bf Bll = ldB(wll + wb);
    v16bf Bah = ldB(wah + wb);
    v16bf Bal = ldB(wal + wb);
#pragma unroll
    for (int ft = 0; ft < 4; ++ft) {
      const __bf16* rh = xh + (ft * 16 + lr) * E_N;
      const __bf16* rl = xl + (ft * 16 + lr) * E_N;
      v16bf Ah = ldA(rh, k0, hf);
      v16bf Al = ldA(rl, k0, hf);
      y1acc[ft] = wmma_bf16(Ah, Blh, y1acc[ft]);
      y1acc[ft] = wmma_bf16(Ah, Bll, y1acc[ft]);
      y1acc[ft] = wmma_bf16(Al, Blh, y1acc[ft]);
      y2acc[ft] = wmma_bf16(Ah, Bah, y2acc[ft]);
      y2acc[ft] = wmma_bf16(Ah, Bal, y2acc[ft]);
      y2acc[ft] = wmma_bf16(Al, Bah, y2acc[ft]);
    }
  }

  // Store y2 transposed ([e][g]) as bf16 hi/lo for stage-2 B fragments.
#pragma unroll
  for (int ft = 0; ft < 4; ++ft) {
#pragma unroll
    for (int r = 0; r < 8; ++r) {
      int gg = ft * 16 + r + 8 * hf;           // row (field) index
      int idx = (e0 + lr) * F_N + gg;          // transposed
      split_bf16(y2acc[ft][r], &y2h[idx], &y2l[idx]);
    }
  }
  __syncthreads();

  // Stage 2: out[b,t] = y1 + b_lin + b_agg + adj[t]@y2   (K = 64, 2 steps)
  const float bsum = b_lin[e0 + lr] + b_agg[e0 + lr];
  for (int t = 0; t < T_N; ++t) {
    const __bf16* ah = adjh + t * (F_N * F_N);
    const __bf16* al = adjl + t * (F_N * F_N);
#pragma unroll
    for (int ft = 0; ft < 4; ++ft) {
      v8f acc = y1acc[ft];
#pragma unroll
      for (int r = 0; r < 8; ++r) acc[r] += bsum;
#pragma unroll
      for (int g0 = 0; g0 < F_N; g0 += 32) {
        v16bf Ah = ldA(ah + (ft * 16 + lr) * F_N, g0, hf);
        v16bf Al = ldA(al + (ft * 16 + lr) * F_N, g0, hf);
        const int yb = (e0 + lr) * F_N + g0 + hf * 16;
        v16bf Bh = ldB(y2h + yb);
        v16bf Bl = ldB(y2l + yb);
        acc = wmma_bf16(Ah, Bh, acc);
        acc = wmma_bf16(Ah, Bl, acc);
        acc = wmma_bf16(Al, Bh, acc);
      }
      float* op = outf + ((size_t)b * T_N + t) * (F_N * E_N);
#pragma unroll
      for (int r = 0; r < 8; ++r)
        op[(ft * 16 + r + 8 * hf) * E_N + e0 + lr] = acc[r];
    }
  }
}

// ---- K2a: deterministic BN partial sums (channels = f*e = 8192, rows = b*t) -
__global__ __launch_bounds__(256) void k_bnpart(const float* __restrict__ outf,
                                                float* __restrict__ psum,
                                                float* __restrict__ psumsq) {
  const int c = blockIdx.x * 256 + threadIdx.x;  // channel 0..8191
  const int r0 = blockIdx.y * 256;               // row chunk of 8192 rows
  const float* p = outf + (size_t)r0 * 8192 + c;
  float s = 0.f, ss = 0.f;
  for (int r = 0; r < 256; ++r) {
    float v = p[(size_t)r * 8192];
    s += v;
    ss += v * v;
  }
  psum[blockIdx.y * 8192 + c] = s;
  psumsq[blockIdx.y * 8192 + c] = ss;
}

// ---- K2b: finalize BN -> fused affine (a, b) per channel --------------------
__global__ __launch_bounds__(256) void k_bnfin(
    const float* __restrict__ psum, const float* __restrict__ psumsq,
    const float* __restrict__ gamma, const float* __restrict__ beta,
    float* __restrict__ bn_a, float* __restrict__ bn_b) {
  const int c = blockIdx.x * 256 + threadIdx.x;
  float s = 0.f, ss = 0.f;
  for (int i = 0; i < 32; ++i) {
    s += psum[i * 8192 + c];
    ss += psumsq[i * 8192 + c];
  }
  float mu = s * (1.f / 8192.f);
  float var = ss * (1.f / 8192.f) - mu * mu;
  float a = gamma[c] * rsqrtf(var + 1e-5f);
  bn_a[c] = a;
  bn_b[c] = beta[c] - mu * a;
}

// ---- K3: BN affine + LayerNorm over (t,f,e) + mean-pool over f --------------
__global__ __launch_bounds__(256) void k_lnpool(
    const float* __restrict__ outf, const float* __restrict__ bn_a,
    const float* __restrict__ bn_b, const float* __restrict__ ln_g,
    const float* __restrict__ ln_b, float* __restrict__ dout) {
  __shared__ float red[256];
  extern __shared__ float lnbuf[];  // 16384 floats (64KB dynamic)
  const int b = blockIdx.x;
  const int tid = threadIdx.x;
  const float* ob = outf + (size_t)b * 16384;

  float vals[64];
  float s = 0.f, ss = 0.f;
#pragma unroll
  for (int i = 0; i < 64; ++i) {
    int j = tid + (i << 8);
    int c = j & 8191;  // channel = f*128+e (same for both towers)
    float v = ob[j] * bn_a[c] + bn_b[c];
    vals[i] = v;
    s += v;
    ss += v * v;
  }
  red[tid] = s;
  __syncthreads();
  for (int st = 128; st > 0; st >>= 1) {
    if (tid < st) red[tid] += red[tid + st];
    __syncthreads();
  }
  const float mean = red[0] * (1.f / 16384.f);
  __syncthreads();
  red[tid] = ss;
  __syncthreads();
  for (int st = 128; st > 0; st >>= 1) {
    if (tid < st) red[tid] += red[tid + st];
    __syncthreads();
  }
  const float var = red[0] * (1.f / 16384.f) - mean * mean;
  const float inv = rsqrtf(var + 1e-5f);
  __syncthreads();
#pragma unroll
  for (int i = 0; i < 64; ++i) {
    int j = tid + (i << 8);
    lnbuf[j] = (vals[i] - mean) * inv * ln_g[j] + ln_b[j];
  }
  __syncthreads();
  // deterministic pool over f: thread tid -> (t = tid>>7, e = tid&127)
  const int t = tid >> 7;
  const int e = tid & 127;
  float acc = 0.f;
#pragma unroll
  for (int f = 0; f < 64; ++f) acc += lnbuf[t * 8192 + f * 128 + e];
  dout[(size_t)b * 256 + tid] = acc * (1.f / 64.f);
}

// ---- launch -----------------------------------------------------------------
extern "C" void kernel_launch(void* const* d_in, const int* in_sizes, int n_in,
                              void* d_out, int out_size, void* d_ws,
                              size_t ws_size, hipStream_t stream) {
  const float* x        = (const float*)d_in[0];
  const float* masker   = (const float*)d_in[1];
  const float* w_lin    = (const float*)d_in[2];
  const float* b_lin    = (const float*)d_in[3];
  const float* w_agg    = (const float*)d_in[4];
  const float* b_agg    = (const float*)d_in[5];
  const float* bn_gamma = (const float*)d_in[6];
  const float* bn_beta  = (const float*)d_in[7];
  const float* ln_gamma = (const float*)d_in[8];
  const float* ln_beta  = (const float*)d_in[9];
  float* out = (float*)d_out;

  // workspace layout (floats)
  float* w      = (float*)d_ws;
  float* adjg   = w;            // 8192
  float* bn_a   = w + 8192;     // 8192
  float* bn_b   = w + 16384;    // 8192
  float* psum   = w + 24576;    // 32*8192
  float* psumsq = w + 286720;   // 32*8192
  float* outf   = w + 548864;   // 4096*2*64*128

  (void)in_sizes; (void)n_in; (void)out_size; (void)ws_size;

  (void)hipFuncSetAttribute((const void*)k_gemm,
                            hipFuncAttributeMaxDynamicSharedMemorySize, 229376);
  (void)hipFuncSetAttribute((const void*)k_lnpool,
                            hipFuncAttributeMaxDynamicSharedMemorySize, 65536);

  k_adj<<<128, 64, 0, stream>>>(masker, adjg);
  k_gemm<<<B_N, 256, 229376, stream>>>(x, w_lin, b_lin, w_agg, b_agg, adjg,
                                       outf);
  k_bnpart<<<dim3(32, 32), 256, 0, stream>>>(outf, psum, psumsq);
  k_bnfin<<<32, 256, 0, stream>>>(psum, psumsq, bn_gamma, bn_beta, bn_a, bn_b);
  k_lnpool<<<B_N, 256, 65536, stream>>>(outf, bn_a, bn_b, ln_gamma, ln_beta,
                                        out);
}
